// TransformerCovarianceModule_51024211476490
// MI455X (gfx1250) — matrix operand order repaired and checked by reading, not compiled, not run on hardware
//
#include <hip/hip_runtime.h>
#include <hip/hip_bf16.h>

typedef __attribute__((ext_vector_type(16))) _Float16 v16h;
typedef __attribute__((ext_vector_type(8)))  _Float16 v8h;
typedef __attribute__((ext_vector_type(8)))  float    v8f;

#define BATCH    2
#define CDIM     256
#define NTOK     2304
#define NHEADS   8
#define HDIM     32
#define QKV_ROWS 768
#define NT       (NTOK / 16)          /* 144 position 16-tiles  */
#define NSTEP    (NTOK / 32)          /* 72  position 32-steps  */
#define SCALE    0.17677669529663687f /* 1/sqrt(32) */

/* Fragment-packed storage: one 16x32 A-tile or 32x16 B-tile = 32 lanes x 16
   halves, stored contiguously as [lane][16] (512 halves = 1 KB).           */

/* A fragment (16M x 32K): lanes 0-15 row M=lane, K kb+0..15 with
   kb=(lane>>4)*16. Inverse map (row,kk): lane=row+16*(kk>>4),
   j=kk&15.                                                   */
/* B fragment (32K x 16N): lanes 0-15 col N=lane, K=0..15; lanes 16-31 col
   N=lane-16, K=16..31. Inverse map (kk,col): lane=col+16*(kk>>4), j=kk&15. */

__device__ __forceinline__ v8f wmma_f16(v16h a, v16h b, v8f c) {
  return __builtin_amdgcn_wmma_f32_16x16x32_f16(false, a, false, b, (short)0, c,
                                                false, false);
}

/* A-fragment loader for row-major f16 (used for Q strip and LDS P strip). */
__device__ __forceinline__ v16h load_a_f16(const _Float16* p0, int lda) {
  const int lane = threadIdx.x & 31;
  const int row  = lane & 15;
  const int kb   = (lane >> 4) << 4;
  const _Float16* p = p0 + row * lda + kb;
  v8h lo = *(const v8h*)(p);
  v8h hi = *(const v8h*)(p + 8);
  v16h a;
#pragma unroll
  for (int j = 0; j < 8; ++j) { a[j] = lo[j]; a[8 + j] = hi[j]; }
  return a;
}

/* ------------- pack kernels: f32 sources -> fragment-packed f16 ---------- */

/* Weights [rows x 256] -> A-frag packed [rows/16][8][lane][16].            */
__global__ __launch_bounds__(128) void pack_a_kernel(
    const float* __restrict__ w, _Float16* __restrict__ wp, int ntiles_o) {
  const int wave = threadIdx.x >> 5, lane = threadIdx.x & 31;
  const int t = blockIdx.x * 4 + wave;              /* t = ot*8 + kt */
  if (t >= ntiles_o * 8) return;
  const int ot = t >> 3, kt = t & 7;
  const int row = ot * 16 + (lane & 15);
  const int kb  = (lane >> 4) << 4;
  const float* src = w + (size_t)row * CDIM + kt * 32 + kb;
  v8h lo, hi;
#pragma unroll
  for (int j = 0; j < 8; ++j) lo[j] = (_Float16)src[j];
#pragma unroll
  for (int j = 0; j < 8; ++j) hi[j] = (_Float16)src[8 + j];
  _Float16* dst = wp + ((size_t)t * 32 + lane) * 16;
  *(v8h*)dst = lo;
  *(v8h*)(dst + 8) = hi;
}

/* x [b][c][n] f32 -> B-frag packed xp[b][kt=c/32][nt=n/16][lane][16].      */
__global__ __launch_bounds__(128) void pack_x_kernel(
    const float* __restrict__ x, _Float16* __restrict__ xp) {
  const int wave = threadIdx.x >> 5, lane = threadIdx.x & 31;
  const int t = blockIdx.x * 4 + wave;              /* t = (b*8+kt)*NT + nt */
  if (t >= BATCH * 8 * NT) return;
  const int nt = t % NT;
  const int bk = t / NT;
  const int b = bk >> 3, kt = bk & 7;
  const int col = nt * 16 + (lane & 15);
  const int kb  = (lane >> 4) << 4;
  const float* src = x + ((size_t)b * CDIM + kt * 32 + kb) * NTOK + col;
  v8h lo, hi;
#pragma unroll
  for (int j = 0; j < 8; ++j) lo[j] = (_Float16)src[(size_t)j * NTOK];
#pragma unroll
  for (int j = 0; j < 8; ++j) hi[j] = (_Float16)src[(size_t)(8 + j) * NTOK];
  _Float16* dst = xp + ((size_t)t * 32 + lane) * 16;
  *(v8h*)dst = lo;
  *(v8h*)(dst + 8) = hi;
}

/* ------------- Kernel 1: QKV projection (packed-fragment GEMM) ----------- */
/* One wave per 16x16 tile of [768 x 2304] per batch; K=256 = 8 WMMA.
   Emits Q row-major [b,h,n,d] (A-frag ready), K/V in B-frag packed form.   */
__global__ __launch_bounds__(128) void qkv_kernel(
    const _Float16* __restrict__ awp, const _Float16* __restrict__ xp,
    const float* __restrict__ b_qkv,
    _Float16* __restrict__ q, _Float16* __restrict__ kp,
    _Float16* __restrict__ vp) {
  const int wave = threadIdx.x >> 5;
  const int lane = threadIdx.x & 31;
  const int tiles_per_b = (QKV_ROWS / 16) * NT;     /* 48*144 */
  const int tile = blockIdx.x * 4 + wave;
  if (tile >= BATCH * tiles_per_b) return;
  const int b  = tile / tiles_per_b;
  const int r  = tile - b * tiles_per_b;
  const int ot = r / NT;
  const int nt = r % NT;
  const int o0 = ot * 16, n0 = nt * 16;

  v8f acc = {};
#pragma unroll
  for (int kt = 0; kt < 8; ++kt) {
    v16h af = *(const v16h*)(awp + (((size_t)ot * 8 + kt) * 32 + lane) * 16);
    v16h bf = *(const v16h*)(xp + ((((size_t)b * 8 + kt) * NT + nt) * 32 + lane) * 16);
    acc = wmma_f16(af, bf, acc);
  }

  const int col   = lane & 15;
  const int mrow0 = (lane >> 4) * 8;
  const int n     = n0 + col;
#pragma unroll
  for (int rr = 0; rr < 8; ++rr) {
    const int o = o0 + mrow0 + rr;
    float val = acc[rr] + b_qkv[o];
    if (o < CDIM) {                      /* Q row-major, softmax scale folded */
      const int h = o >> 5, d = o & 31;
      q[(((size_t)b * NHEADS + h) * NTOK + n) * HDIM + d] = (_Float16)(val * SCALE);
    } else if (o < 2 * CDIM) {           /* K -> B-frag packed per key tile  */
      const int c = o - CDIM, h = c >> 5, d = c & 31;
      const int mt = n >> 4;
      const int lt = (n & 15) + ((d >> 4) << 4);
      kp[((((size_t)b * NHEADS + h) * NT + mt) * 32 + lt) * 16 + (d & 15)] =
          (_Float16)val;
    } else {                             /* V -> B-frag packed per 32-step   */
      const int c = o - 2 * CDIM, h = c >> 5, d = c & 31;
      const int step = n >> 5, mm = n & 31;
      const int dt = d >> 4;
      const int lt = (d & 15) + ((mm >> 4) << 4);
      vp[(((((size_t)b * NHEADS + h) * NSTEP + step) * 2 + dt) * 32 + lt) * 16 +
         (mm & 15)] = (_Float16)val;
    }
  }
}

/* ------------- Kernel 2: fused attention per (b, h, 16-query tile) ------- */
#define SMEM_S_BYTES (16 * NTOK * 4)   /* f32 scores  147456 */
#define SMEM_P_BYTES (16 * NTOK * 2)   /* f16 probs    73728 */
#define SMEM_R_BYTES (8 * 256 * 4)     /* split-K part  8192 */
#define SMEM_TOTAL   (SMEM_S_BYTES + SMEM_P_BYTES + SMEM_R_BYTES) /* <320K   */

__global__ __launch_bounds__(256) void attn_kernel(
    const _Float16* __restrict__ q, const _Float16* __restrict__ kp,
    const _Float16* __restrict__ vp, _Float16* __restrict__ yp) {
  extern __shared__ char smem[];
  float*    sS = (float*)smem;
  _Float16* sP = (_Float16*)(smem + SMEM_S_BYTES);
  float*    sR = (float*)(smem + SMEM_S_BYTES + SMEM_P_BYTES);

  const int wave = threadIdx.x >> 5;
  const int lane = threadIdx.x & 31;
  const int qt   = blockIdx.x % NT;
  const int h    = (blockIdx.x / NT) % NHEADS;
  const int b    = blockIdx.x / (NT * NHEADS);
  const int q0   = qt * 16;

  const _Float16* qbase = q + (((size_t)b * NHEADS + h) * NTOK + q0) * HDIM;
  const _Float16* kpb   = kp + ((size_t)b * NHEADS + h) * NT * 512;
  const _Float16* vpb   = vp + ((size_t)b * NHEADS + h) * NSTEP * 2 * 512;

  const int col   = lane & 15;
  const int mrow0 = (lane >> 4) * 8;

  /* Phase 1: S(16 x 2304) = Q_tile * K ; d=32 -> one v_wmma per key tile.
     B fragment = single contiguous 32B load per lane from packed K.        */
  v16h aq = load_a_f16(qbase, HDIM);
  for (int mt = wave; mt < NT; mt += 8) {
    const _Float16* kt = kpb + (size_t)mt * 512 + lane * 16;
    if (mt + 8 < NT) __builtin_prefetch(kpb + (size_t)(mt + 8) * 512 + lane * 16, 0, 0);
    v16h bk = *(const v16h*)kt;
    v8f s = {};
    s = wmma_f16(aq, bk, s);
#pragma unroll
    for (int rr = 0; rr < 8; ++rr)
      sS[(mrow0 + rr) * NTOK + mt * 16 + col] = s[rr];
  }
  __syncthreads();

  /* Phase 2: row softmax; one wave per 2 rows, wave32 shfl_xor reductions  */
#pragma unroll
  for (int rw = 0; rw < 2; ++rw) {
    const int row = wave * 2 + rw;
    float* sr = sS + row * NTOK;
    float m = -3.4e38f;
    for (int i = lane; i < NTOK; i += 32) m = fmaxf(m, sr[i]);
#pragma unroll
    for (int off = 16; off >= 1; off >>= 1) m = fmaxf(m, __shfl_xor(m, off, 32));
    float sum = 0.f;
    for (int i = lane; i < NTOK; i += 32) {
      float e = __expf(sr[i] - m);
      sr[i] = e;
      sum += e;
    }
#pragma unroll
    for (int off = 16; off >= 1; off >>= 1) sum += __shfl_xor(sum, off, 32);
    const float inv = 1.0f / sum;
    _Float16* pr = sP + row * NTOK;
    for (int i = lane; i < NTOK; i += 32) pr[i] = (_Float16)(sr[i] * inv);
  }
  __syncthreads();

  /* Phase 3: O(16 x 32) = P * V^T ; 2 d-tiles x split-K over 4 waves each.
     A from LDS P strip (ds b128 loads), B = 32B contiguous from packed V.  */
  const int dtile = wave & 1;
  const int kc    = wave >> 1;
  v8f acc = {};
  for (int step = kc * 18; step < kc * 18 + 18; ++step) {
    v16h ap = load_a_f16(sP + step * 32, NTOK);
    v16h bv = *(const v16h*)(vpb + (((size_t)step * 2 + dtile) * 32 + lane) * 16);
    acc = wmma_f16(ap, bv, acc);
  }
#pragma unroll
  for (int rr = 0; rr < 8; ++rr)
    sR[wave * 256 + (mrow0 + rr) * 16 + col] = acc[rr];
  __syncthreads();

  /* Reduce split-K partials; scatter into B-frag packed yp for projection. */
  if (wave < 2) {
#pragma unroll
    for (int rr = 0; rr < 8; ++rr) {
      const int idx = (mrow0 + rr) * 16 + col;
      float o = sR[wave * 256 + idx] + sR[(wave + 2) * 256 + idx] +
                sR[(wave + 4) * 256 + idx] + sR[(wave + 6) * 256 + idx];
      const int c = h * HDIM + wave * 16 + col;    /* channel  */
      const int n = q0 + mrow0 + rr;               /* position */
      const int kt = c >> 5, sub = c & 31;
      const int nt = n >> 4;
      const int lt = (n & 15) + ((sub >> 4) << 4);
      yp[((((size_t)b * 8 + kt) * NT + nt) * 32 + lt) * 16 + (sub & 15)] =
          (_Float16)o;
    }
  }
}

/* ------------- Kernel 3: projection + bias + residual (packed GEMM) ------ */
__global__ __launch_bounds__(128) void proj_kernel(
    const _Float16* __restrict__ yp, const _Float16* __restrict__ pwp,
    const float* __restrict__ b_proj, const float* __restrict__ x,
    float* __restrict__ out) {
  const int wave = threadIdx.x >> 5;
  const int lane = threadIdx.x & 31;
  const int tiles_per_b = (CDIM / 16) * NT;        /* 16*144 */
  const int tile = blockIdx.x * 4 + wave;
  if (tile >= BATCH * tiles_per_b) return;
  const int b  = tile / tiles_per_b;
  const int r  = tile - b * tiles_per_b;
  const int ot = r / NT;
  const int nt = r % NT;
  const int o0 = ot * 16, n0 = nt * 16;

  v8f acc = {};
#pragma unroll
  for (int kt = 0; kt < 8; ++kt) {
    v16h af = *(const v16h*)(pwp + (((size_t)ot * 8 + kt) * 32 + lane) * 16);
    v16h bf = *(const v16h*)(yp + ((((size_t)b * 8 + kt) * NT + nt) * 32 + lane) * 16);
    acc = wmma_f16(af, bf, acc);
  }

  const int col   = lane & 15;
  const int mrow0 = (lane >> 4) * 8;
#pragma unroll
  for (int rr = 0; rr < 8; ++rr) {
    const int o = o0 + mrow0 + rr;
    const size_t idx = ((size_t)b * CDIM + o) * NTOK + n0 + col;
    out[idx] = acc[rr] + b_proj[o] + x[idx];
  }
}

/* ------------- Host launcher --------------------------------------------- */
extern "C" void kernel_launch(void* const* d_in, const int* in_sizes, int n_in,
                              void* d_out, int out_size, void* d_ws, size_t ws_size,
                              hipStream_t stream) {
  (void)in_sizes; (void)n_in; (void)out_size; (void)ws_size;
  const float* x      = (const float*)d_in[0];
  const float* w_qkv  = (const float*)d_in[1];
  const float* b_qkv  = (const float*)d_in[2];
  const float* w_proj = (const float*)d_in[3];
  const float* b_proj = (const float*)d_in[4];
  float* out = (float*)d_out;

  const size_t qkv_elems = (size_t)BATCH * NHEADS * NTOK * HDIM; /* 1,179,648 */
  _Float16* q   = (_Float16*)d_ws;
  _Float16* kp  = q   + qkv_elems;
  _Float16* vp  = kp  + qkv_elems;
  _Float16* yp  = vp  + qkv_elems;
  _Float16* xp  = yp  + qkv_elems;
  _Float16* awp = xp  + qkv_elems;                 /* 48*8*512 halves */
  _Float16* pwp = awp + (size_t)48 * 8 * 512;      /* 16*8*512 halves */

  pack_a_kernel<<<(48 * 8) / 4, 128, 0, stream>>>(w_qkv, awp, 48);
  pack_a_kernel<<<(16 * 8) / 4, 128, 0, stream>>>(w_proj, pwp, 16);
  pack_x_kernel<<<(BATCH * 8 * NT) / 4, 128, 0, stream>>>(x, xp);

  qkv_kernel<<<(BATCH * (QKV_ROWS / 16) * NT) / 4, 128, 0, stream>>>(
      awp, xp, b_qkv, q, kp, vp);

  (void)hipFuncSetAttribute(reinterpret_cast<const void*>(attn_kernel),
                            hipFuncAttributeMaxDynamicSharedMemorySize, SMEM_TOTAL);
  attn_kernel<<<BATCH * NHEADS * NT, 256, SMEM_TOTAL, stream>>>(q, kp, vp, yp);

  proj_kernel<<<(BATCH * (CDIM / 16) * NT) / 4, 128, 0, stream>>>(
      yp, pwp, b_proj, x, out);
}
